// ConvMambaBlock_21921513079114
// MI455X (gfx1250) — matrix operand order repaired and checked by reading, not compiled, not run on hardware
//
#include <hip/hip_runtime.h>
#include <hip/hip_bf16.h>
#include <math.h>

// ---------------------------------------------------------------------------
// ConvMamba block for MI455X (gfx1250, wave32, WMMA).
// GEMMs: v_wmma_f32_16x16x32_f16, register-blocked 32x64 macro-tile per wave
//        (MT=2 x NT=4 accumulators, A fragment reused across N tiles).
// Scan:  one wave per (batch, channel); lane = SSM state (32 states = wave32).
// ---------------------------------------------------------------------------

#define DIM 256
#define D_STATE 32
#define E_DIM 512
#define DT_RANK 16
#define L_SEQ 1024
#define B_SZ 4
#define NTOK (B_SZ * L_SEQ)   // 4096

typedef __attribute__((ext_vector_type(16))) _Float16 v16h;
typedef __attribute__((ext_vector_type(8)))  float    v8f;

union HV {
    v16h v;
    uint4 q[2];
};

__device__ __forceinline__ float silu_f(float x) {
    return x / (1.0f + __expf(-x));
}
__device__ __forceinline__ float softplus_f(float x) {
    return fmaxf(x, 0.0f) + log1pf(__expf(-fabsf(x)));
}
__device__ __forceinline__ float gelu_f(float x) {
    return 0.5f * x * (1.0f + erff(x * 0.70710678118654752f));
}

// ----------------------------- small elementwise ---------------------------

__global__ void k_f32_to_f16(const float* __restrict__ s, _Float16* __restrict__ d, int n) {
    int i = blockIdx.x * blockDim.x + threadIdx.x;
    if (i < n) d[i] = (_Float16)s[i];
}

// dt_proj_w (E, 16) -> padded f16 (E, 32)
__global__ void k_pad_dtw(const float* __restrict__ s, _Float16* __restrict__ d) {
    int i = blockIdx.x * blockDim.x + threadIdx.x;
    if (i >= E_DIM * 32) return;
    int k = i & 31, e = i >> 5;
    d[i] = (k < DT_RANK) ? (_Float16)s[e * DT_RANK + k] : (_Float16)0.0f;
}

// dbl (T,80) cols 0..15 -> padded f16 (T, 32)
__global__ void k_pad_dtr(const float* __restrict__ dbl, _Float16* __restrict__ d) {
    int i = blockIdx.x * blockDim.x + threadIdx.x;
    if (i >= NTOK * 32) return;
    int k = i & 31, t = i >> 5;
    d[i] = (k < DT_RANK) ? (_Float16)dbl[t * 80 + k] : (_Float16)0.0f;
}

// ------------------------------ layernorm ----------------------------------
// one wave per token (DIM=256 -> 8 floats/lane), wave32 shfl butterfly reduce
__global__ void k_layernorm(const float* __restrict__ x,
                            const float* __restrict__ g, const float* __restrict__ b,
                            float* __restrict__ of32, _Float16* __restrict__ of16,
                            int ntok) {
    int wid  = (blockIdx.x * blockDim.x + threadIdx.x) >> 5;
    int lane = threadIdx.x & 31;
    if (wid >= ntok) return;
    const float* xr = x + (size_t)wid * DIM;
    float4 v0 = *(const float4*)(xr + lane * 8);
    float4 v1 = *(const float4*)(xr + lane * 8 + 4);
    float s  = v0.x + v0.y + v0.z + v0.w + v1.x + v1.y + v1.z + v1.w;
    float s2 = v0.x*v0.x + v0.y*v0.y + v0.z*v0.z + v0.w*v0.w +
               v1.x*v1.x + v1.y*v1.y + v1.z*v1.z + v1.w*v1.w;
    #pragma unroll
    for (int o = 16; o; o >>= 1) {
        s  += __shfl_xor(s, o, 32);
        s2 += __shfl_xor(s2, o, 32);
    }
    float mean = s * (1.0f / DIM);
    float var  = s2 * (1.0f / DIM) - mean * mean;
    float rinv = rsqrtf(var + 1e-5f);
    float4 g0 = *(const float4*)(g + lane * 8);
    float4 g1 = *(const float4*)(g + lane * 8 + 4);
    float4 b0 = *(const float4*)(b + lane * 8);
    float4 b1 = *(const float4*)(b + lane * 8 + 4);
    float in[8]  = {v0.x, v0.y, v0.z, v0.w, v1.x, v1.y, v1.z, v1.w};
    float gg[8]  = {g0.x, g0.y, g0.z, g0.w, g1.x, g1.y, g1.z, g1.w};
    float bb[8]  = {b0.x, b0.y, b0.z, b0.w, b1.x, b1.y, b1.z, b1.w};
    size_t base = (size_t)wid * DIM + lane * 8;
    #pragma unroll
    for (int j = 0; j < 8; ++j) {
        float o = (in[j] - mean) * rinv * gg[j] + bb[j];
        if (of32) of32[base + j] = o;
        if (of16) of16[base + j] = (_Float16)o;
    }
}

// -------------------- dwconv k=3 same-pad (+h) -> f16 ----------------------
__global__ void k_conv1(const float* __restrict__ h, const float* __restrict__ w,
                        const float* __restrict__ bias, _Float16* __restrict__ out) {
    int i = blockIdx.x * blockDim.x + threadIdx.x;
    if (i >= B_SZ * L_SEQ * DIM) return;
    int c = i & (DIM - 1);
    int l = (i >> 8) & (L_SEQ - 1);
    int b = i >> 18;
    const float* hr = h + ((size_t)b * L_SEQ) * DIM + c;
    float acc = bias[c];
    #pragma unroll
    for (int j = 0; j < 3; ++j) {
        int ll = l - 1 + j;
        if (ll >= 0 && ll < L_SEQ) acc += w[c * 3 + j] * hr[(size_t)ll * DIM];
    }
    out[i] = (_Float16)(acc + hr[(size_t)l * DIM]);
}

// -------- causal dwconv k=4 + bias + SiLU over xin part of xz --------------
__global__ void k_ssmconv(const float* __restrict__ xz, const float* __restrict__ w,
                          const float* __restrict__ bias,
                          float* __restrict__ u, _Float16* __restrict__ uh) {
    int i = blockIdx.x * blockDim.x + threadIdx.x;
    if (i >= B_SZ * L_SEQ * E_DIM) return;
    int e = i & (E_DIM - 1);
    int l = (i >> 9) & (L_SEQ - 1);
    int b = i >> 19;
    const float* xr = xz + ((size_t)b * L_SEQ) * (2 * E_DIM) + e;
    float acc = bias[e];
    #pragma unroll
    for (int j = 0; j < 4; ++j) {
        int ll = l - 3 + j;
        if (ll >= 0) acc += w[e * 4 + j] * xr[(size_t)ll * (2 * E_DIM)];
    }
    float uv = silu_f(acc);
    u[i]  = uv;
    uh[i] = (_Float16)uv;
}

// --------------------------- selective scan --------------------------------
// one wave per (b, e); lane n = SSM state n (D_STATE == 32 == wave32)
__global__ void k_scan(const float* __restrict__ u, const float* __restrict__ dt,
                       const float* __restrict__ A_log, const float* __restrict__ dbl,
                       const float* __restrict__ Dp, float* __restrict__ y) {
    int wid  = (blockIdx.x * blockDim.x + threadIdx.x) >> 5;
    int lane = threadIdx.x & 31;
    if (wid >= B_SZ * E_DIM) return;
    int b = wid >> 9;           // / E_DIM
    int e = wid & (E_DIM - 1);
    float a  = -__expf(A_log[e * D_STATE + lane]);
    float Dv = Dp[e];
    float h = 0.0f;
    size_t t0 = (size_t)b * L_SEQ;
    for (int l = 0; l < L_SEQ; ++l) {
        size_t t = t0 + l;
        float dtv = dt[t * E_DIM + e];
        float uv  = u[t * E_DIM + e];
        float Bn  = dbl[t * 80 + DT_RANK + lane];
        float Cn  = dbl[t * 80 + DT_RANK + D_STATE + lane];
        float dA  = __expf(dtv * a);
        h = dA * h + dtv * Bn * uv;
        float p = h * Cn;
        #pragma unroll
        for (int o = 16; o; o >>= 1) p += __shfl_xor(p, o, 32);
        if (lane == 0) y[t * E_DIM + e] = p + uv * Dv;
    }
}

// ------------------------- y * silu(z) -> f16 ------------------------------
__global__ void k_yact(const float* __restrict__ y, const float* __restrict__ xz,
                       _Float16* __restrict__ yh) {
    int i = blockIdx.x * blockDim.x + threadIdx.x;
    if (i >= NTOK * E_DIM) return;
    int e = i & (E_DIM - 1);
    int t = i >> 9;
    float z = xz[(size_t)t * (2 * E_DIM) + E_DIM + e];
    yh[i] = (_Float16)(y[i] * silu_f(z));
}

// ------------------------------ WMMA GEMM ----------------------------------
// C[M,N] = A[M,K] * W[N,K]^T ; (16*MT)x(16*NT) macro-tile per wave.
// A fragments reused across NT tiles; B fragments reused across MT tiles.
// MODE 0: C=acc            MODE 1: C=acc+resid
// MODE 2: C=softplus(acc+bias)
// MODE 3: Ch=f16(gelu(acc+bias))   MODE 4: C=acc+bias+resid
template <int MODE, int MT, int NT>
__global__ void k_gemm(const _Float16* __restrict__ A, int lda,
                       const _Float16* __restrict__ Bw, int ldb,
                       float* __restrict__ C, _Float16* __restrict__ Ch, int ldc,
                       int M, int N, int K,
                       const float* __restrict__ bias,
                       const float* __restrict__ resid) {
    int wave = (blockIdx.x * blockDim.x + threadIdx.x) >> 5;
    int lane = threadIdx.x & 31;
    int ntile = N / (16 * NT);
    int total = (M / (16 * MT)) * ntile;
    if (wave >= total) return;
    int tm = (wave / ntile) * (16 * MT);
    int tn = (wave % ntile) * (16 * NT);

    int hlf = lane >> 4;        // 0 or 1
    int l16 = lane & 15;
    const _Float16* Arow[MT];
    #pragma unroll
    for (int m = 0; m < MT; ++m)
        Arow[m] = A + (size_t)(tm + m * 16 + l16) * lda;
    const _Float16* Bcol[NT];
    #pragma unroll
    for (int t = 0; t < NT; ++t)
        Bcol[t] = Bw + (size_t)(tn + t * 16 + l16) * ldb;

    v8f acc[MT][NT] = {};
    for (int k0 = 0; k0 < K; k0 += 32) {
        HV a[MT];
        // A 16x32 f16: lane<16 -> K [k0,k0+8)+[k0+16,k0+24); lane>=16 shifted by 8
        #pragma unroll
        for (int m = 0; m < MT; ++m) {
            a[m].q[0] = *(const uint4*)(Arow[m] + k0 + hlf * 8);
            a[m].q[1] = *(const uint4*)(Arow[m] + k0 + 16 + hlf * 8);
        }
        #pragma unroll
        for (int t = 0; t < NT; ++t) {
            // B 32x16 f16: lane<16 -> K [k0,k0+8)+[k0+16,k0+24) of col n; lane>=16 shifted by 8
            HV b;
            b.q[0] = *(const uint4*)(Bcol[t] + k0 + hlf * 8);
            b.q[1] = *(const uint4*)(Bcol[t] + k0 + 16 + hlf * 8);
            #pragma unroll
            for (int m = 0; m < MT; ++m)
                acc[m][t] = __builtin_amdgcn_wmma_f32_16x16x32_f16(
                    false, a[m].v, false, b.v, (short)0, acc[m][t], false, false);
        }
    }

    #pragma unroll
    for (int m = 0; m < MT; ++m) {
        int row0 = tm + m * 16 + hlf * 8;
        #pragma unroll
        for (int t = 0; t < NT; ++t) {
            int col = tn + t * 16 + l16;
            size_t cb = (size_t)row0 * ldc + col;
            float bv = (MODE == 2 || MODE == 3 || MODE == 4) ? bias[col] : 0.0f;
            #pragma unroll
            for (int r = 0; r < 8; ++r) {
                float v = acc[m][t][r];
                size_t idx = cb + (size_t)r * ldc;
                if (MODE == 0) {
                    C[idx] = v;
                } else if (MODE == 1) {
                    C[idx] = v + resid[idx];
                } else if (MODE == 2) {
                    C[idx] = softplus_f(v + bv);
                } else if (MODE == 3) {
                    Ch[idx] = (_Float16)gelu_f(v + bv);
                } else if (MODE == 4) {
                    C[idx] = v + bv + resid[idx];
                }
            }
        }
    }
}

// ---------------------------------------------------------------------------

static inline int ceil_div(int a, int b) { return (a + b - 1) / b; }

extern "C" void kernel_launch(void* const* d_in, const int* in_sizes, int n_in,
                              void* d_out, int out_size, void* d_ws, size_t ws_size,
                              hipStream_t stream) {
    const float* x       = (const float*)d_in[0];
    const float* n1g     = (const float*)d_in[1];
    const float* n1b     = (const float*)d_in[2];
    const float* conv_w  = (const float*)d_in[3];
    const float* conv_b  = (const float*)d_in[4];
    const float* in_w    = (const float*)d_in[5];
    const float* sconv_w = (const float*)d_in[6];
    const float* sconv_b = (const float*)d_in[7];
    const float* xp_w    = (const float*)d_in[8];
    const float* dt_w    = (const float*)d_in[9];
    const float* dt_b    = (const float*)d_in[10];
    const float* A_log   = (const float*)d_in[11];
    const float* Dp      = (const float*)d_in[12];
    const float* out_w   = (const float*)d_in[13];
    const float* n2g     = (const float*)d_in[14];
    const float* n2b     = (const float*)d_in[15];
    const float* m1w     = (const float*)d_in[16];
    const float* m1b     = (const float*)d_in[17];
    const float* m2w     = (const float*)d_in[18];
    const float* m2b     = (const float*)d_in[19];
    float* out = (float*)d_out;

    // ---- workspace layout (256B aligned slabs) ----
    char* ws = (char*)d_ws;
    size_t off = 0;
    auto alloc = [&](size_t bytes) -> char* {
        char* p = ws + off;
        off = (off + bytes + 255) & ~(size_t)255;
        return p;
    };
    float*    h_ln1  = (float*)   alloc((size_t)NTOK * DIM * 4);
    _Float16* min_h  = (_Float16*)alloc((size_t)NTOK * DIM * 2);
    _Float16* w_in_h = (_Float16*)alloc((size_t)2 * E_DIM * DIM * 2);
    float*    xz     = (float*)   alloc((size_t)NTOK * 2 * E_DIM * 4);
    float*    u      = (float*)   alloc((size_t)NTOK * E_DIM * 4);
    _Float16* u_h    = (_Float16*)alloc((size_t)NTOK * E_DIM * 2);
    _Float16* w_xp_h = (_Float16*)alloc((size_t)80 * E_DIM * 2);
    float*    dbl    = (float*)   alloc((size_t)NTOK * 80 * 4);
    _Float16* dtr_h  = (_Float16*)alloc((size_t)NTOK * 32 * 2);
    _Float16* w_dt_h = (_Float16*)alloc((size_t)E_DIM * 32 * 2);
    float*    dt     = (float*)   alloc((size_t)NTOK * E_DIM * 4);
    float*    y      = (float*)   alloc((size_t)NTOK * E_DIM * 4);
    _Float16* yact_h = (_Float16*)alloc((size_t)NTOK * E_DIM * 2);
    _Float16* w_ou_h = (_Float16*)alloc((size_t)DIM * E_DIM * 2);
    float*    h2     = (float*)   alloc((size_t)NTOK * DIM * 4);
    _Float16* mln_h  = (_Float16*)alloc((size_t)NTOK * DIM * 2);
    _Float16* w_m1_h = (_Float16*)alloc((size_t)4 * DIM * DIM * 2);
    _Float16* hid_h  = (_Float16*)alloc((size_t)NTOK * 4 * DIM * 2);
    _Float16* w_m2_h = (_Float16*)alloc((size_t)DIM * 4 * DIM * 2);
    (void)ws_size;

    // ---- weight conversions to f16 ----
    auto cvt = [&](const float* s, _Float16* d, int n) {
        k_f32_to_f16<<<ceil_div(n, 256), 256, 0, stream>>>(s, d, n);
    };
    cvt(in_w,  w_in_h, 2 * E_DIM * DIM);
    cvt(xp_w,  w_xp_h, 80 * E_DIM);
    cvt(out_w, w_ou_h, DIM * E_DIM);
    cvt(m1w,   w_m1_h, 4 * DIM * DIM);
    cvt(m2w,   w_m2_h, DIM * 4 * DIM);
    k_pad_dtw<<<ceil_div(E_DIM * 32, 256), 256, 0, stream>>>(dt_w, w_dt_h);

    // ---- 1. layernorm1 ----
    k_layernorm<<<NTOK / 8, 256, 0, stream>>>(x, n1g, n1b, h_ln1, (_Float16*)nullptr, NTOK);

    // ---- 2. dwconv3 + h -> f16 ----
    k_conv1<<<ceil_div(NTOK * DIM, 256), 256, 0, stream>>>(h_ln1, conv_w, conv_b, min_h);

    // ---- 3. in_proj GEMM: (4096,256)x(256,1024) -> xz ----
    {
        int waves = (NTOK / 32) * (2 * E_DIM / 64);   // 2048
        k_gemm<0, 2, 4><<<waves / 8, 256, 0, stream>>>(min_h, DIM, w_in_h, DIM,
            xz, nullptr, 2 * E_DIM, NTOK, 2 * E_DIM, DIM, nullptr, nullptr);
    }

    // ---- 4. causal dwconv4 + SiLU -> u ----
    k_ssmconv<<<ceil_div(NTOK * E_DIM, 256), 256, 0, stream>>>(xz, sconv_w, sconv_b, u, u_h);

    // ---- 5. x_proj GEMM: (4096,512)x(512,80) -> dbl (NT=5 covers N=80) ----
    {
        int waves = (NTOK / 32) * 1;                  // 128
        k_gemm<0, 2, 5><<<waves / 8, 256, 0, stream>>>(u_h, E_DIM, w_xp_h, E_DIM,
            dbl, nullptr, 80, NTOK, 80, E_DIM, nullptr, nullptr);
    }

    // ---- 6. dt_proj GEMM (K padded to 32) + bias + softplus -> dt ----
    k_pad_dtr<<<ceil_div(NTOK * 32, 256), 256, 0, stream>>>(dbl, dtr_h);
    {
        int waves = (NTOK / 32) * (E_DIM / 64);       // 1024
        k_gemm<2, 2, 4><<<waves / 8, 256, 0, stream>>>(dtr_h, 32, w_dt_h, 32,
            dt, nullptr, E_DIM, NTOK, E_DIM, 32, dt_b, nullptr);
    }

    // ---- 7. selective scan ----
    k_scan<<<(B_SZ * E_DIM) / 8, 256, 0, stream>>>(u, dt, A_log, dbl, Dp, y);

    // ---- 8. y * silu(z) -> f16 ----
    k_yact<<<ceil_div(NTOK * E_DIM, 256), 256, 0, stream>>>(y, xz, yact_h);

    // ---- 9. out_proj GEMM + residual(x) -> h2 ----
    {
        int waves = (NTOK / 32) * (DIM / 64);         // 512
        k_gemm<1, 2, 4><<<waves / 8, 256, 0, stream>>>(yact_h, E_DIM, w_ou_h, E_DIM,
            h2, nullptr, DIM, NTOK, DIM, E_DIM, nullptr, x);
    }

    // ---- 10. layernorm2 -> f16 ----
    k_layernorm<<<NTOK / 8, 256, 0, stream>>>(h2, n2g, n2b, (float*)nullptr, mln_h, NTOK);

    // ---- 11. MLP1 GEMM + bias + GELU -> f16 hidden ----
    {
        int waves = (NTOK / 32) * (4 * DIM / 64);     // 2048
        k_gemm<3, 2, 4><<<waves / 8, 256, 0, stream>>>(mln_h, DIM, w_m1_h, DIM,
            nullptr, hid_h, 4 * DIM, NTOK, 4 * DIM, DIM, m1b, nullptr);
    }

    // ---- 12. MLP2 GEMM + bias + residual(h2) -> out ----
    {
        int waves = (NTOK / 32) * (DIM / 64);         // 512
        k_gemm<4, 2, 4><<<waves / 8, 256, 0, stream>>>(hid_h, 4 * DIM, w_m2_h, 4 * DIM,
            out, nullptr, DIM, NTOK, DIM, 4 * DIM, m2b, h2);
    }
}
